// VectorQuantizedVAE2_30391188586510
// MI455X (gfx1250) — matrix operand order repaired and checked by reading, not compiled, not run on hardware
//
#include <hip/hip_runtime.h>
#include <hip/hip_bf16.h>
#include <cstdint>

// ---------------------------------------------------------------------------
// VQ-VAE decoder forward for MI455X (gfx1250, wave32, WMMA 16x16x32 f16).
// - All matmul-shaped ops on v_wmma_f32_16x16x32_f16 (fp32 accumulate).
// - conv3x3 halo tile staged via GLOBAL_LOAD_ASYNC_TO_LDS_B128 (ASYNCcnt).
// - conv1x1 weight tile staged via the Tensor Data Mover (TENSORcnt).
// ---------------------------------------------------------------------------

typedef __attribute__((ext_vector_type(16))) _Float16 v16h;
typedef __attribute__((ext_vector_type(8)))  _Float16 v8h;
typedef __attribute__((ext_vector_type(8)))  float    v8f;
typedef __attribute__((ext_vector_type(4)))  unsigned int u32x4;
typedef __attribute__((ext_vector_type(8)))  int      i32x8;
typedef __attribute__((ext_vector_type(4)))  int      i32x4;

#define DEV __device__ __forceinline__

DEV v8f wmma_f16(v16h a, v16h b, v8f c) {
  // 8 args: (neg_a, A, neg_b, B, c_mod, C, reuse_a, reuse_b)
  return __builtin_amdgcn_wmma_f32_16x16x32_f16(false, a, false, b, (short)0, c, false, false);
}

// A fragment (16x32 f16): lane L holds row M=L%16; K contiguous in
// [(L/16)*16, +16)  (matches B fragment order).
DEV v16h a_frag(const _Float16* rowbase, int k0, int half) {
  const _Float16* p = rowbase + k0 + (half << 4);
  v8h lo = *(const v8h*)p;
  v8h hi = *(const v8h*)(p + 8);
  v16h r;
#pragma unroll
  for (int i = 0; i < 8; ++i) { r[i] = lo[i]; r[i + 8] = hi[i]; }
  return r;
}

DEV v16h zfrag() {
  v16h r;
#pragma unroll
  for (int i = 0; i < 16; ++i) r[i] = (_Float16)0.0f;
  return r;
}

DEV v8f zacc() {
  v8f a;
#pragma unroll
  for (int i = 0; i < 8; ++i) a[i] = 0.0f;
  return a;
}

// TDM: 1-D contiguous tile, global -> LDS. D# groups per ISA 8.3/8.4:
// g0 = {count=1 | type=2(image), lds_addr, global_addr lo, global_addr hi}
// g1 = {data_size=8B, tensor_dim0, tensor_dim1=1, tile_dim0, tile_dim1=1,
//       tensor_dim0_stride, ...}; groups 2/3 zero (<=2D tensor).
DEV void tdm_load_1d_64k(unsigned lds_addr, unsigned long long gaddr) {
  u32x4 g0;
  g0[0] = 1u;                                    // count=1, user descriptor
  g0[1] = lds_addr;                              // lds_addr [63:32]
  g0[2] = (unsigned)gaddr;                       // global_addr [95:64]
  g0[3] = (unsigned)(gaddr >> 32) | (2u << 30);  // addr hi | type=2
  i32x8 g1;
  g1[0] = (3 << 16);                 // wg_mask=0, data_size=3 (8B units)
  g1[1] = (int)(8192u << 16);        // tensor_dim0 = 8192 (lo16 @ [31:16])
  g1[2] = (1 << 16);                 // tensor_dim0 hi=0 | tensor_dim1 lo=1
  g1[3] = (int)(8192u << 16);        // tensor_dim1 hi=0 | tile_dim0 = 8192
  g1[4] = 1;                         // tile_dim1 = 1 | tile_dim2 = 0
  g1[5] = 8192;                      // tensor_dim0_stride lo32
  g1[6] = 0;
  g1[7] = 0;
  i32x4 gz = {0, 0, 0, 0};
#if defined(__clang_major__) && (__clang_major__ >= 23)
  i32x8 z8 = {0, 0, 0, 0, 0, 0, 0, 0};
  __builtin_amdgcn_tensor_load_to_lds(g0, g1, gz, gz, z8, 0);
#else
  __builtin_amdgcn_tensor_load_to_lds(g0, g1, gz, gz, 0);
#endif
}

// ----------------------------- prep / pack ---------------------------------

// z_e_x NCHW fp32 [32,256,32,32] -> NHWC f16 [32768,256]
__global__ void k_prep_z(const float* __restrict__ z, _Float16* __restrict__ zf) {
  int e = blockIdx.x * 256 + threadIdx.x;      // < 8388608
  int c = e & 255, pix = e >> 8;
  int b = pix >> 10, yx = pix & 1023;
  zf[e] = (_Float16)z[(((b << 8) + c) << 10) + yx];
}

// codebook fp32 [512,256] -> f16 [512,256] + per-row |c|^2
__global__ void k_pack_cb(const float* __restrict__ cb, _Float16* __restrict__ cb16,
                          float* __restrict__ cnorm) {
  __shared__ float red[256];
  int r = blockIdx.x, t = threadIdx.x;
  float v = cb[r * 256 + t];
  cb16[r * 256 + t] = (_Float16)v;
  red[t] = v * v;
  __syncthreads();
#pragma unroll
  for (int s = 128; s > 0; s >>= 1) {
    if (t < s) red[t] += red[t + s];
    __syncthreads();
  }
  if (t == 0) cnorm[r] = red[0];
}

// conv3x3 weights OIHW (256,256,3,3) fp32 -> [tap][O][I] f16
__global__ void k_packw3(const float* __restrict__ w, _Float16* __restrict__ wp) {
  int e = blockIdx.x * 256 + threadIdx.x;      // < 589824
  int tap = e >> 16, o = (e >> 8) & 255, i = e & 255;
  int ky = tap / 3, kx = tap % 3;
  wp[e] = (_Float16)w[((((o << 8) + i) * 3 + ky) * 3 + kx)];
}

// generic fp32 -> f16 cast (conv1x1 weights already [O][I])
__global__ void k_cast16(const float* __restrict__ s, _Float16* __restrict__ d, int n) {
  int e = blockIdx.x * 256 + threadIdx.x;
  if (e < n) d[e] = (_Float16)s[e];
}

// deconv1 weights (in=256,out=256,4,4) fp32 -> [ky*4+kx][O][I] f16
__global__ void k_packdw1(const float* __restrict__ w, _Float16* __restrict__ wp) {
  int e = blockIdx.x * 256 + threadIdx.x;      // < 1048576
  int kk = e >> 16, o = (e >> 8) & 255, i = e & 255;
  int ky = kk >> 2, kx = kk & 3;
  wp[e] = (_Float16)w[(((i << 8) + o) << 4) + (ky << 2) + kx];
}

// deconv2 weights (in=256,out=3,4,4) fp32 -> [ky*4+kx][16 padded O][I] f16
__global__ void k_packdw2(const float* __restrict__ w, _Float16* __restrict__ wp) {
  int e = blockIdx.x * 256 + threadIdx.x;      // < 65536
  int kk = e >> 12, o = (e >> 8) & 15, i = e & 255;
  int ky = kk >> 2, kx = kk & 3;
  wp[e] = (o < 3) ? (_Float16)w[((i * 3 + o) << 4) + (ky << 2) + kx] : (_Float16)0.0f;
}

__global__ void k_zero(float* __restrict__ p, int n) {
  int e = blockIdx.x * 256 + threadIdx.x;
  if (e < n) p[e] = 0.0f;
}

// ------------------------------- VQ ----------------------------------------

// per wave: 16 rows of z, loop all K=512 codes in 16-col tiles, 8 k-steps each.
// argmin of (|c|^2 - 2 z.c); |z|^2 is row-constant and dropped.
__global__ void k_vq(const _Float16* __restrict__ zf, const _Float16* __restrict__ cb16,
                     const float* __restrict__ cnorm, int* __restrict__ idx) {
  int tid = threadIdx.x, wave = tid >> 5, lane = tid & 31;
  int m = lane & 15, half = lane >> 4;
  int row0 = blockIdx.x * 64 + wave * 16;
  const _Float16* arow = zf + ((row0 + m) << 8);
  float bv[8];
  int bi[8];
#pragma unroll
  for (int r = 0; r < 8; ++r) { bv[r] = 3.0e38f; bi[r] = 0; }
  for (int n0 = 0; n0 < 512; n0 += 16) {
    v8f acc = zacc();
    const _Float16* brow = cb16 + ((n0 + m) << 8);
#pragma unroll
    for (int ks = 0; ks < 8; ++ks) {
      v16h a = a_frag(arow, ks * 32, half);
      v16h bb = *(const v16h*)(brow + ks * 32 + (half << 4));
      acc = wmma_f16(a, bb, acc);
    }
    float cn = cnorm[n0 + m];
#pragma unroll
    for (int r = 0; r < 8; ++r) {
      float v = cn - 2.0f * acc[r];
      if (v < bv[r]) { bv[r] = v; bi[r] = n0 + m; }
    }
  }
  // reduce over the 16 lanes that share each row (first-min tie-break on idx)
#pragma unroll
  for (int r = 0; r < 8; ++r) {
    float v = bv[r];
    int i = bi[r];
    for (int s = 8; s >= 1; s >>= 1) {
      float ov = __shfl_xor(v, s, 16);
      int oi = __shfl_xor(i, s, 16);
      if (ov < v || (ov == v && oi < i)) { v = ov; i = oi; }
    }
    if (m == 0) idx[row0 + r + (half << 3)] = i;
  }
}

// gather codes: h0 fp32 NHWC, relu(code) f16 NHWC (rb1 input), z_q_x NCHW fp32
__global__ void k_gather(const int* __restrict__ idx, const float* __restrict__ cb,
                         float* __restrict__ h0, _Float16* __restrict__ t16,
                         float* __restrict__ zq) {
  int e = blockIdx.x * 256 + threadIdx.x;      // < 8388608
  int c = e & 255, pix = e >> 8;
  float v = cb[(idx[pix] << 8) + c];
  h0[e] = v;
  t16[e] = (_Float16)(v > 0.0f ? v : 0.0f);
  int b = pix >> 10, yx = pix & 1023;
  zq[(((b << 8) + c) << 10) + yx] = v;
}

// ------------------------------ convs --------------------------------------

// 3x3 conv, pad 1, NHWC [32768,256] f16 -> fp32. Halo tile (3x18 pixels x 256ch)
// staged in LDS with async global->LDS copies: pre-zero LDS (border padding),
// per-lane masked global_load_async_to_lds_b128, s_wait_asynccnt, barrier.
__global__ void k_conv3(const _Float16* __restrict__ in16, const _Float16* __restrict__ wp,
                        const float* __restrict__ bias, float* __restrict__ out) {
  __shared__ __align__(32) _Float16 sm[3 * 18 * 256];
  int tid = threadIdx.x;
  int p0 = blockIdx.x * 16;
  int b = p0 >> 10, y = (p0 >> 5) & 31, x0 = p0 & 31;
  // zero-fill (covers zero padding at tensor borders)
  for (int i = tid; i < 3 * 18 * 32; i += 256) {
    v8h z;
#pragma unroll
    for (int j = 0; j < 8; ++j) z[j] = (_Float16)0.0f;
    *(v8h*)(sm + i * 8) = z;
  }
  __syncthreads();
  // async-copy the in-bounds portion (16B chunks), ASYNCcnt-tracked
  for (int i = tid; i < 3 * 18 * 32; i += 256) {
    int ci  = (i & 31) << 3;        // halves within pixel
    int xi  = (i >> 5) % 18;
    int dyi = (i >> 5) / 18;
    int gy = y + dyi - 1, gx = x0 + xi - 1;
    if (gy >= 0 && gy < 32 && gx >= 0 && gx < 32) {
      unsigned lds = (unsigned)(uintptr_t)(sm + (dyi * 18 + xi) * 256 + ci);
      unsigned long long ga = (unsigned long long)(uintptr_t)
          (in16 + ((((b << 5) + gy) << 5) + gx) * 256 + ci);
      asm volatile("global_load_async_to_lds_b128 %0, %1, off"
                   :: "v"(lds), "v"(ga) : "memory");
    }
  }
  asm volatile("s_wait_asynccnt 0x0" ::: "memory");
  __syncthreads();

  int wave = tid >> 5, lane = tid & 31, m = lane & 15, half = lane >> 4;
  int n = blockIdx.y * 128 + wave * 16 + m;
  v8f acc = zacc();
  for (int tap = 0; tap < 9; ++tap) {
    int dy = tap / 3, dx = tap % 3;
    const _Float16* arow = sm + (dy * 18 + m + dx) * 256;
    const _Float16* brow = wp + ((tap << 8) + n) * 256;
    __builtin_prefetch(brow + 256, 0, 3);   // -> global_prefetch_b8
#pragma unroll
    for (int ks = 0; ks < 8; ++ks) {
      v16h a = a_frag(arow, ks * 32, half);
      v16h bb = *(const v16h*)(brow + ks * 32 + (half << 4));
      acc = wmma_f16(a, bb, acc);
    }
  }
  float bn = bias[n];
#pragma unroll
  for (int r = 0; r < 8; ++r)
    out[((p0 + r + (half << 3)) << 8) + n] = acc[r] + bn;
}

// 1x1 conv: pure GEMM [32768,256]x[256,256]. The per-block weight slice
// ([128 outCh][256 K] f16 = 64KB contiguous) is DMA'd into LDS by the
// Tensor Data Mover (one wave issues, TENSORcnt wait, then barrier).
__global__ void k_conv1(const _Float16* __restrict__ in16, const _Float16* __restrict__ wp,
                        const float* __restrict__ bias, float* __restrict__ out) {
  __shared__ __align__(128) _Float16 smb[128 * 256];   // 64KB weight tile
  int tid = threadIdx.x, wave = tid >> 5, lane = tid & 31, m = lane & 15, half = lane >> 4;
  if (wave == 0) {
    unsigned lds = (unsigned)(uintptr_t)smb;
    unsigned long long ga = (unsigned long long)(uintptr_t)
        (wp + ((size_t)blockIdx.y << 15));             // 128*256 halves per slice
    tdm_load_1d_64k(lds, ga);
    __builtin_amdgcn_s_wait_tensorcnt(0);
  }
  __syncthreads();

  int p0 = blockIdx.x * 16;
  int nl = (wave << 4) + m;                            // 0..127 within slice
  int n = blockIdx.y * 128 + nl;
  const _Float16* arow = in16 + ((p0 + m) << 8);
  const _Float16* brow = smb + (nl << 8);
  v8f acc = zacc();
#pragma unroll
  for (int ks = 0; ks < 8; ++ks) {
    v16h a = a_frag(arow, ks * 32, half);
    v16h bb = *(const v16h*)(brow + ks * 32 + (half << 4));
    acc = wmma_f16(a, bb, acc);
  }
  float bn = bias[n];
#pragma unroll
  for (int r = 0; r < 8; ++r)
    out[((p0 + r + (half << 3)) << 8) + n] = acc[r] + bn;
}

// ----------------------------- batchnorm -----------------------------------

__global__ void k_bnstats(const float* __restrict__ x, float* __restrict__ stats) {
  int c = threadIdx.x;
  int p0 = blockIdx.x * 64;
  float s = 0.0f, s2 = 0.0f;
  for (int i = 0; i < 64; ++i) {
    float v = x[(p0 + i) * 256 + c];
    s += v;
    s2 += v * v;
  }
  atomicAdd(&stats[c], s);
  atomicAdd(&stats[256 + c], s2);
}

__global__ void k_bn_relu16(const float* __restrict__ x, const float* __restrict__ stats,
                            const float* __restrict__ g, const float* __restrict__ be,
                            _Float16* __restrict__ y, float invn) {
  int e = blockIdx.x * 256 + threadIdx.x;
  int c = e & 255;
  float mean = stats[c] * invn;
  float var = fmaxf(stats[256 + c] * invn - mean * mean, 0.0f);
  float sc = g[c] * rsqrtf(var + 1e-5f);
  float v = (x[e] - mean) * sc + be[c];
  y[e] = (_Float16)(v > 0.0f ? v : 0.0f);
}

__global__ void k_bn_add(const float* __restrict__ x, const float* __restrict__ stats,
                         const float* __restrict__ g, const float* __restrict__ be,
                         float* __restrict__ h, float invn) {
  int e = blockIdx.x * 256 + threadIdx.x;
  int c = e & 255;
  float mean = stats[c] * invn;
  float var = fmaxf(stats[256 + c] * invn - mean * mean, 0.0f);
  float sc = g[c] * rsqrtf(var + 1e-5f);
  h[e] += (x[e] - mean) * sc + be[c];
}

__global__ void k_relu16(const float* __restrict__ x, _Float16* __restrict__ y) {
  int e = blockIdx.x * 256 + threadIdx.x;
  float v = x[e];
  y[e] = (_Float16)(v > 0.0f ? v : 0.0f);
}

// ----------------------------- deconvs -------------------------------------

// ConvTranspose2d k=4,s=2,p=1, 256->256, 32x32 -> 64x64. Tiles grouped by
// output-x parity so each 16-pixel tile shares a fixed 2x2 tap set.
__global__ void k_deconv1(const _Float16* __restrict__ in16, const _Float16* __restrict__ wp,
                          const float* __restrict__ bias, float* __restrict__ out) {
  int tid = threadIdx.x, wave = tid >> 5, lane = tid & 31, m = lane & 15, half = lane >> 4;
  int mt = blockIdx.x;
  int b = mt >> 8, rem = mt & 255;
  int oy = rem >> 2, p = (rem >> 1) & 1, t = rem & 1;
  int n = blockIdx.y * 128 + wave * 16 + m;
  v8f acc = zacc();
  int ky0 = (oy + 1) & 1, kx0 = (p + 1) & 1;
#pragma unroll
  for (int kyi = 0; kyi < 2; ++kyi) {
    int ky = ky0 + 2 * kyi;
    int iy = (oy + 1 - ky) >> 1;
    if (iy < 0 || iy >= 32) continue;
#pragma unroll
    for (int kxi = 0; kxi < 2; ++kxi) {
      int kx = kx0 + 2 * kxi;
      int ix = (((p + 1 - kx) >> 1) + t * 16) + m;
      bool aval = (ix >= 0 && ix < 32);
      const _Float16* arow = in16 + ((((b << 5) + iy) << 5) + ix) * 256;
      const _Float16* brow = wp + ((((ky << 2) + kx) << 8) + n) * 256;
      __builtin_prefetch(brow, 0, 3);
#pragma unroll
      for (int ks = 0; ks < 8; ++ks) {
        v16h a = aval ? a_frag(arow, ks * 32, half) : zfrag();
        v16h bb = *(const v16h*)(brow + ks * 32 + (half << 4));
        acc = wmma_f16(a, bb, acc);
      }
    }
  }
  float bn = bias[n];
#pragma unroll
  for (int r = 0; r < 8; ++r) {
    int mrow = r + (half << 3);
    int ox = p + 2 * (t * 16 + mrow);
    out[((((b << 6) + oy) << 6) + ox) * 256 + n] = acc[r] + bn;
  }
}

// ConvTranspose2d k=4,s=2,p=1, 256->3 (padded to 16 cols), 64x64 -> 128x128,
// fused bias + tanh, writes x_tilde NCHW directly. 8 waves = 8 M-tiles/block.
__global__ void k_deconv2(const _Float16* __restrict__ in16, const _Float16* __restrict__ wp,
                          const float* __restrict__ bias, float* __restrict__ out) {
  int tid = threadIdx.x, wave = tid >> 5, lane = tid & 31, m = lane & 15, half = lane >> 4;
  int mt = blockIdx.x * 8 + wave;
  int b = mt >> 10, rem = mt & 1023;
  int oy = rem >> 3, p = (rem >> 2) & 1, t = rem & 3;
  v8f acc = zacc();
  int ky0 = (oy + 1) & 1, kx0 = (p + 1) & 1;
#pragma unroll
  for (int kyi = 0; kyi < 2; ++kyi) {
    int ky = ky0 + 2 * kyi;
    int iy = (oy + 1 - ky) >> 1;
    if (iy < 0 || iy >= 64) continue;
#pragma unroll
    for (int kxi = 0; kxi < 2; ++kxi) {
      int kx = kx0 + 2 * kxi;
      int ix = (((p + 1 - kx) >> 1) + t * 16) + m;
      bool aval = (ix >= 0 && ix < 64);
      const _Float16* arow = in16 + ((((b << 6) + iy) << 6) + ix) * 256;
      const _Float16* brow = wp + ((((ky << 2) + kx) << 4) + m) * 256;
#pragma unroll
      for (int ks = 0; ks < 8; ++ks) {
        v16h a = aval ? a_frag(arow, ks * 32, half) : zfrag();
        v16h bb = *(const v16h*)(brow + ks * 32 + (half << 4));
        acc = wmma_f16(a, bb, acc);
      }
    }
  }
  if (m < 3) {
    float bn = bias[m];
#pragma unroll
    for (int r = 0; r < 8; ++r) {
      int mrow = r + (half << 3);
      int ox = p + 2 * (t * 16 + mrow);
      out[((b * 3 + m) * 128 + oy) * 128 + ox] = tanhf(acc[r] + bn);
    }
  }
}

// ---------------------------------------------------------------------------

extern "C" void kernel_launch(void* const* d_in, const int* in_sizes, int n_in,
                              void* d_out, int out_size, void* d_ws, size_t ws_size,
                              hipStream_t stream) {
  (void)in_sizes; (void)n_in; (void)out_size; (void)ws_size;
  const float* z_e_x = (const float*)d_in[0];
  const float* codebook = (const float*)d_in[1];
  const float* rb1_w1 = (const float*)d_in[2];
  const float* rb1_b1 = (const float*)d_in[3];
  const float* rb1_g1 = (const float*)d_in[4];
  const float* rb1_be1 = (const float*)d_in[5];
  const float* rb1_w2 = (const float*)d_in[6];
  const float* rb1_b2 = (const float*)d_in[7];
  const float* rb1_g2 = (const float*)d_in[8];
  const float* rb1_be2 = (const float*)d_in[9];
  const float* rb2_w1 = (const float*)d_in[10];
  const float* rb2_b1 = (const float*)d_in[11];
  const float* rb2_g1 = (const float*)d_in[12];
  const float* rb2_be1 = (const float*)d_in[13];
  const float* rb2_w2 = (const float*)d_in[14];
  const float* rb2_b2 = (const float*)d_in[15];
  const float* rb2_g2 = (const float*)d_in[16];
  const float* rb2_be2 = (const float*)d_in[17];
  const float* dc1_w = (const float*)d_in[18];
  const float* dc1_b = (const float*)d_in[19];
  const float* bn3_g = (const float*)d_in[20];
  const float* bn3_b = (const float*)d_in[21];
  const float* dc2_w = (const float*)d_in[22];
  const float* dc2_b = (const float*)d_in[23];

  float* out = (float*)d_out;
  float* out_ze = out + 1572864;   // 32*3*128*128
  float* out_zq = out + 9961472;   // + 32*256*32*32

  // ---- workspace layout (bytes, 256-aligned) ----
  char* ws = (char*)d_ws;
  size_t o = 0;
  auto alloc = [&](size_t bytes) { size_t r = o; o = (o + bytes + 255) & ~(size_t)255; return r; };
  size_t o_stats = alloc(512 * 4);
  size_t o_cnorm = alloc(512 * 4);
  size_t o_idx   = alloc(32768 * 4);
  size_t o_cb16  = alloc(512 * 256 * 2);
  size_t o_wp3a  = alloc(9 * 256 * 256 * 2);
  size_t o_wp3b  = alloc(9 * 256 * 256 * 2);
  size_t o_wp1a  = alloc(256 * 256 * 2);
  size_t o_wp1b  = alloc(256 * 256 * 2);
  size_t o_wpd1  = alloc(16 * 256 * 256 * 2);
  size_t o_wpd2  = alloc(16 * 16 * 256 * 2);
  size_t o_t16   = alloc((size_t)32768 * 256 * 2);   // f16 act 32x32 (also zf16)
  size_t o_t16b  = alloc((size_t)32768 * 256 * 2);
  size_t o_h0    = alloc((size_t)32768 * 256 * 4);
  size_t o_c1    = alloc((size_t)32768 * 256 * 4);
  size_t o_c2    = alloc((size_t)32768 * 256 * 4);   // c1+c2 contiguous -> d116 alias
  size_t o_d1    = alloc((size_t)131072 * 256 * 4);

  float*    stats = (float*)(ws + o_stats);
  float*    cnorm = (float*)(ws + o_cnorm);
  int*      idx   = (int*)(ws + o_idx);
  _Float16* cb16  = (_Float16*)(ws + o_cb16);
  _Float16* wp3a  = (_Float16*)(ws + o_wp3a);
  _Float16* wp3b  = (_Float16*)(ws + o_wp3b);
  _Float16* wp1a  = (_Float16*)(ws + o_wp1a);
  _Float16* wp1b  = (_Float16*)(ws + o_wp1b);
  _Float16* wpd1  = (_Float16*)(ws + o_wpd1);
  _Float16* wpd2  = (_Float16*)(ws + o_wpd2);
  _Float16* t16   = (_Float16*)(ws + o_t16);
  _Float16* t16b  = (_Float16*)(ws + o_t16b);
  float*    h0    = (float*)(ws + o_h0);
  float*    c1    = (float*)(ws + o_c1);
  float*    c2    = (float*)(ws + o_c2);
  float*    d1    = (float*)(ws + o_d1);
  _Float16* d116  = (_Float16*)(ws + o_c1);          // 64MB alias over c1+c2

  // ---- packing / prep ----
  k_pack_cb<<<512, 256, 0, stream>>>(codebook, cb16, cnorm);
  k_prep_z<<<32768, 256, 0, stream>>>(z_e_x, t16);   // t16 <- z (zf16 role)
  k_packw3<<<2304, 256, 0, stream>>>(rb1_w1, wp3a);
  k_packw3<<<2304, 256, 0, stream>>>(rb2_w1, wp3b);
  k_cast16<<<256, 256, 0, stream>>>(rb1_w2, wp1a, 65536);
  k_cast16<<<256, 256, 0, stream>>>(rb2_w2, wp1b, 65536);
  k_packdw1<<<4096, 256, 0, stream>>>(dc1_w, wpd1);
  k_packdw2<<<256, 256, 0, stream>>>(dc2_w, wpd2);
  hipMemcpyAsync(out_ze, z_e_x, (size_t)8388608 * 4, hipMemcpyDeviceToDevice, stream);

  // ---- vector quantization ----
  k_vq<<<512, 128, 0, stream>>>(t16, cb16, cnorm, idx);
  k_gather<<<32768, 256, 0, stream>>>(idx, codebook, h0, t16, out_zq); // t16 <- relu(codes)

  // ---- resblock 1 ----
  k_conv3<<<dim3(2048, 2), 256, 0, stream>>>(t16, wp3a, rb1_b1, c1);
  k_zero<<<2, 256, 0, stream>>>(stats, 512);
  k_bnstats<<<512, 256, 0, stream>>>(c1, stats);
  k_bn_relu16<<<32768, 256, 0, stream>>>(c1, stats, rb1_g1, rb1_be1, t16b, 1.0f / 32768.0f);
  k_conv1<<<dim3(2048, 2), 256, 0, stream>>>(t16b, wp1a, rb1_b2, c2);
  k_zero<<<2, 256, 0, stream>>>(stats, 512);
  k_bnstats<<<512, 256, 0, stream>>>(c2, stats);
  k_bn_add<<<32768, 256, 0, stream>>>(c2, stats, rb1_g2, rb1_be2, h0, 1.0f / 32768.0f);

  // ---- resblock 2 ----
  k_relu16<<<32768, 256, 0, stream>>>(h0, t16);
  k_conv3<<<dim3(2048, 2), 256, 0, stream>>>(t16, wp3b, rb2_b1, c1);
  k_zero<<<2, 256, 0, stream>>>(stats, 512);
  k_bnstats<<<512, 256, 0, stream>>>(c1, stats);
  k_bn_relu16<<<32768, 256, 0, stream>>>(c1, stats, rb2_g1, rb2_be1, t16b, 1.0f / 32768.0f);
  k_conv1<<<dim3(2048, 2), 256, 0, stream>>>(t16b, wp1b, rb2_b2, c2);
  k_zero<<<2, 256, 0, stream>>>(stats, 512);
  k_bnstats<<<512, 256, 0, stream>>>(c2, stats);
  k_bn_add<<<32768, 256, 0, stream>>>(c2, stats, rb2_g2, rb2_be2, h0, 1.0f / 32768.0f);

  // ---- decoder tail ----
  k_relu16<<<32768, 256, 0, stream>>>(h0, t16);
  k_deconv1<<<dim3(8192, 2), 256, 0, stream>>>(t16, wpd1, dc1_b, d1);
  k_zero<<<2, 256, 0, stream>>>(stats, 512);
  k_bnstats<<<2048, 256, 0, stream>>>(d1, stats);
  k_bn_relu16<<<131072, 256, 0, stream>>>(d1, stats, bn3_g, bn3_b, d116, 1.0f / 131072.0f);
  k_deconv2<<<4096, 256, 0, stream>>>(d116, wpd2, dc2_b, out);
}
